// BiMambaBlock_38560216383844
// MI455X (gfx1250) — compile-verified, matrix operand order checked by reading, not run on hardware
//
#include <hip/hip_runtime.h>
#include <hip/hip_bf16.h>

// ---------------------------------------------------------------------------
// BiMamba block for MI455X (gfx1250, wave32, WMMA + async-LDS + TDM).
// GEMMs: bf16 WMMA, 128x128 block tile, double-buffered LDS staging:
//   A tile via per-lane global_load_async_to_lds_b128 (ASYNCcnt),
//   B tile via one Tensor Data Mover descriptor (tensor_load_to_lds,
//   TENSORcnt), fragments consumed via ds_load_b128.
// Scan: sequential selective scan, B/C staged per-timestep with async b32.
// ---------------------------------------------------------------------------

typedef __attribute__((ext_vector_type(16))) __bf16 v16bf;
typedef __attribute__((ext_vector_type(8)))  __bf16 v8bf;
typedef __attribute__((ext_vector_type(8)))  float  v8f;
typedef __attribute__((ext_vector_type(4)))  unsigned u32x4;
typedef __attribute__((ext_vector_type(8)))  unsigned u32x8;

union FragAB { v16bf v; v8bf h[2]; };

// ------------------------------ helpers ------------------------------------

__device__ __forceinline__ void async_ld_b128(unsigned lds_off, const void* gp) {
    unsigned long long ga = (unsigned long long)gp;
    asm volatile("global_load_async_to_lds_b128 %0, %1, off"
                 :: "v"(lds_off), "v"(ga) : "memory");
}

__device__ __forceinline__ void async_ld_b32(unsigned lds_off, const void* gp) {
    unsigned long long ga = (unsigned long long)gp;
    asm volatile("global_load_async_to_lds_b32 %0, %1, off"
                 :: "v"(lds_off), "v"(ga) : "memory");
}

// One 2D TDM tile load: bf16 elements, tile_d0 x tile_d1 (K-major rows),
// row stride = stride_elems. All inputs wave-uniform -> D# lives in SGPRs.
__device__ __forceinline__ void tdm_load_2d(unsigned lds_addr, const void* gaddr,
                                            unsigned tensor_d0, unsigned tensor_d1,
                                            unsigned stride_elems,
                                            unsigned tile_d0, unsigned tile_d1) {
    unsigned long long ga = (unsigned long long)gaddr;
    u32x4 g0 = { 1u,                                   // count=1, user mode
                 lds_addr,                             // lds_addr [63:32]
                 (unsigned)ga,                         // global_addr lo
                 (unsigned)(ga >> 32) | (2u << 30) };  // addr hi | type=2
    u32x8 g1 = { 1u << 16,                             // data_size=1 (2 bytes)
                 (tensor_d0 & 0xffffu) << 16,          // tensor_dim0 lo16
                 (tensor_d0 >> 16) | ((tensor_d1 & 0xffffu) << 16),
                 (tensor_d1 >> 16) | (tile_d0 << 16),  // | tile_dim0
                 tile_d1,                              // tile_dim1 (tile_dim2=0)
                 stride_elems,                         // tensor_dim0_stride lo32
                 0u, 0u };
    asm volatile("tensor_load_to_lds %0, %1" :: "s"(g0), "s"(g1) : "memory");
}

__device__ __forceinline__ float block_reduce_sum256(float v) {
    __shared__ float sm[8];
    #pragma unroll
    for (int m = 16; m >= 1; m >>= 1) v += __shfl_xor(v, m, 32);
    const int lane = threadIdx.x & 31;
    const int w    = threadIdx.x >> 5;
    if (lane == 0) sm[w] = v;
    __syncthreads();
    float r = sm[lane & 7];
    #pragma unroll
    for (int m = 4; m >= 1; m >>= 1) r += __shfl_xor(r, m, 32);
    __syncthreads();
    return r;
}

__device__ __forceinline__ float silu_f(float x) {
    return x / (1.0f + __expf(-x));
}

// -------------------- weight conversion (f32 -> bf16, row pad) -------------

__global__ __launch_bounds__(256)
void k_cvt_pad_bf16(const float* __restrict__ src, __bf16* __restrict__ dst,
                    int rows_src, int rows_dst, int cols) {
    int i = blockIdx.x * 256 + threadIdx.x;
    int total = rows_dst * cols;
    if (i >= total) return;
    int r = i / cols, c = i % cols;
    dst[i] = (r < rows_src) ? (__bf16)src[(size_t)r * cols + c] : (__bf16)0.0f;
}

// -------------------- input RMSNorm (writes fwd + time-flipped) ------------

__global__ __launch_bounds__(256)
void k_rmsnorm_in(const float* __restrict__ x, const float* __restrict__ w,
                  __bf16* __restrict__ xn, __bf16* __restrict__ xn_rev) {
    const int row = blockIdx.x;                 // 0..4095 = b*2048 + t
    const float* xr = x + (size_t)row * 512;
    const int c0 = threadIdx.x, c1 = threadIdx.x + 256;
    float v0 = xr[c0], v1 = xr[c1];
    float ss = block_reduce_sum256(v0 * v0 + v1 * v1);
    float rs = rsqrtf(ss * (1.0f / 512.0f) + 1e-5f);
    const int bb = row >> 11, t = row & 2047;
    const int rrev = (bb << 11) + (2047 - t);
    __bf16 o0 = (__bf16)(v0 * rs * w[c0]);
    __bf16 o1 = (__bf16)(v1 * rs * w[c1]);
    xn[(size_t)row * 512 + c0] = o0;
    xn[(size_t)row * 512 + c1] = o1;
    xn_rev[(size_t)rrev * 512 + c0] = o0;
    xn_rev[(size_t)rrev * 512 + c1] = o1;
}

// -------------------- WMMA GEMM: C = A(MxK) * B(NxK)^T ---------------------
// A, B bf16 row-major (K contiguous), B rows padded so n-index never OOB.
// Block tile 128x128, K-step 32, double-buffered staging:
//   A: per-lane async b128, B: one TDM descriptor per stage (wave 0).
// MODE 0: fp32 store to Cf (guard col < N)
// MODE 1: bf16 store to Cb at [maybe-flipped row, col_off + col]
// MODE 2: fp32 store to Cf with +bias[col] +resid[row*ldc+col]

template <int MODE>
__global__ __launch_bounds__(256)
void k_gemm_wmma(const __bf16* __restrict__ A, const __bf16* __restrict__ B,
                 int M, int N, int K, int Brows,
                 float* __restrict__ Cf, __bf16* __restrict__ Cb, int ldc,
                 int col_off, int flip,
                 const float* __restrict__ bias, const float* __restrict__ resid) {
    // LDS tiles: 128 rows x 32 K-elements bf16, row stride 64B, double buffered
    __shared__ __bf16 smA[2][128 * 32];
    __shared__ __bf16 smB[2][128 * 32];

    const int tid  = threadIdx.x;
    const int lane = tid & 31;
    const int wave = tid >> 5;                   // 8 waves
    const int wm = wave & 3, wn = wave >> 2;     // 4 x 2 wave grid
    const int m0 = blockIdx.y * 128;
    const int n0 = blockIdx.x * 128;
    const int rA = lane & 15;
    const int kh = (lane >> 4) << 3;             // 0 or 8 (K half select)

    // A staging map: thread covers one 16B segment; two issues cover 128 rows
    const int srow = tid >> 2;                   // 0..63
    const int sseg = tid & 3;                    // 16B segment within 64B row
    const unsigned ldsA[2] = { (unsigned)(uintptr_t)&smA[0][0],
                               (unsigned)(uintptr_t)&smA[1][0] };
    const unsigned ldsB[2] = { (unsigned)(uintptr_t)&smB[0][0],
                               (unsigned)(uintptr_t)&smB[1][0] };

    v8f c[2][4];
    #pragma unroll
    for (int mi = 0; mi < 2; ++mi)
        #pragma unroll
        for (int ni = 0; ni < 4; ++ni)
            c[mi][ni] = (v8f)(0.0f);

    auto stage = [&](int buf, int kk) {
        #pragma unroll
        for (int j = 0; j < 2; ++j) {            // A tile: async per-lane
            const int r = j * 64 + srow;
            async_ld_b128(ldsA[buf] + (unsigned)(r * 64 + (sseg << 4)),
                          A + (size_t)(m0 + r) * K + kk + (sseg << 3));
        }
        if (wave == 0)                           // B tile: one TDM descriptor
            tdm_load_2d(ldsB[buf], B + (size_t)n0 * K + kk,
                        (unsigned)K, (unsigned)Brows, (unsigned)K, 32u, 128u);
    };

    auto compute = [&](int buf) {
        FragAB a[2], b[4];
        #pragma unroll
        for (int mi = 0; mi < 2; ++mi) {
            const __bf16* p = &smA[buf][(wm * 32 + mi * 16 + rA) * 32 + kh];
            a[mi].h[0] = *(const v8bf*)(p);
            a[mi].h[1] = *(const v8bf*)(p + 16);
        }
        #pragma unroll
        for (int ni = 0; ni < 4; ++ni) {
            const __bf16* p = &smB[buf][(wn * 64 + ni * 16 + rA) * 32 + kh];
            b[ni].h[0] = *(const v8bf*)(p);
            b[ni].h[1] = *(const v8bf*)(p + 16);
        }
        #pragma unroll
        for (int mi = 0; mi < 2; ++mi)
            #pragma unroll
            for (int ni = 0; ni < 4; ++ni)
                c[mi][ni] = __builtin_amdgcn_wmma_f32_16x16x32_bf16(
                    false, a[mi].v, false, b[ni].v,
                    (short)0, c[mi][ni], false, false);
    };

    const int nIter = K >> 5;
    stage(0, 0);
    for (int it = 0; it + 1 < nIter; ++it) {     // steady state: no branches
        stage((it & 1) ^ 1, (it + 1) << 5);
        asm volatile("s_wait_asynccnt 0x2" ::: "memory");
        __builtin_amdgcn_s_wait_tensorcnt(1);
        __syncthreads();                         // publish LDS writes
        compute(it & 1);
        __syncthreads();                         // reads done before rewrite
    }
    asm volatile("s_wait_asynccnt 0x0" ::: "memory");
    __builtin_amdgcn_s_wait_tensorcnt(0);
    __syncthreads();
    compute((nIter - 1) & 1);                    // peeled last iteration

    // Epilogue: C/D layout — VGPR r: lanes0-15 => M=r, lanes16-31 => M=r+8,
    // N = lane & 15.
    const int rowAdd = (lane >> 4) << 3;
    const int colLn  = lane & 15;
    #pragma unroll
    for (int mi = 0; mi < 2; ++mi) {
        #pragma unroll
        for (int ni = 0; ni < 4; ++ni) {
            #pragma unroll
            for (int r = 0; r < 8; ++r) {
                int row = m0 + wm * 32 + mi * 16 + r + rowAdd;
                int col = n0 + wn * 64 + ni * 16 + colLn;
                float v = c[mi][ni][r];
                if (MODE == 0) {
                    if (col < N) Cf[(size_t)row * ldc + col] = v;
                } else if (MODE == 1) {
                    int bb = row >> 11, t = row & 2047;
                    int row2 = flip ? (bb << 11) + (2047 - t) : row;
                    Cb[(size_t)row2 * ldc + col_off + col] = (__bf16)v;
                } else {
                    Cf[(size_t)row * ldc + col] =
                        v + bias[col] + resid[(size_t)row * ldc + col];
                }
            }
        }
    }
}

// -------------------- depthwise causal conv(4) + SiLU ----------------------
// reads zxbcdt cols [1024,2176) , writes xBC (4096 x 1152)

__global__ __launch_bounds__(256)
void k_conv_silu(const float* __restrict__ zx, const float* __restrict__ cw,
                 const float* __restrict__ cb, float* __restrict__ xBC) {
    int i = blockIdx.x * 256 + threadIdx.x;
    if (i >= 4096 * 1152) return;
    int row = i / 1152, ch = i % 1152;
    int t = row & 2047, rb = row - t;           // rb = batch base row
    float acc = cb[ch];
    #pragma unroll
    for (int j = 0; j < 4; ++j) {
        int ti = t - 3 + j;
        if (ti >= 0)
            acc += cw[ch * 4 + j] * zx[(size_t)(rb + ti) * 2192 + 1024 + ch];
    }
    xBC[(size_t)row * 1152 + ch] = silu_f(acc);
}

// -------------------- selective state-space scan ---------------------------
// one block per (batch, head); 256 threads; thread = (p = tid>>2, nq = tid&3)
// owning 16 of 64 state columns. h kept in registers, B/C async-staged in LDS.

__global__ __launch_bounds__(256)
void k_scan(const float* __restrict__ zx, const float* __restrict__ xBC,
            float* __restrict__ y,
            const float* __restrict__ dt_bias, const float* __restrict__ A_log,
            const float* __restrict__ Dp) {
    const int head = blockIdx.x & 15;
    const int bb   = blockIdx.x >> 4;
    const int tid  = threadIdx.x;
    const int p    = tid >> 2;
    const int nq   = tid & 3;
    __shared__ float sBC[128];                   // [0,64)=B, [64,128)=C
    const unsigned ldsBC = (unsigned)(uintptr_t)&sBC[0];

    float h[16];
    #pragma unroll
    for (int i = 0; i < 16; ++i) h[i] = 0.0f;

    const float dtb  = dt_bias[head];
    const float negA = -__expf(A_log[head]);
    const float Dh   = Dp[head];
    const size_t rz0 = (size_t)(bb * 2048) * 2192;
    const size_t rx0 = (size_t)(bb * 2048) * 1152;
    const int nb = nq * 16;

    for (int t = 0; t < 2048; ++t) {
        const float* zr = zx  + rz0 + (size_t)t * 2192;
        const float* xr = xBC + rx0 + (size_t)t * 1152;
        if (tid < 128)                            // B at 1024, C at 1088
            async_ld_b32(ldsBC + tid * 4, xr + 1024 + tid);
        asm volatile("s_wait_asynccnt 0x0" ::: "memory");
        __syncthreads();

        float dtr = zr[2176 + head] + dtb;
        float dt  = (dtr > 20.0f) ? dtr : log1pf(__expf(dtr));   // softplus
        float dA  = __expf(dt * negA);
        float xp  = xr[head * 64 + p];
        float dtx = dt * xp;

        float part = 0.0f;
        #pragma unroll
        for (int i = 0; i < 16; ++i) {
            float hv = dA * h[i] + dtx * sBC[nb + i];
            h[i] = hv;
            part += hv * sBC[64 + nb + i];
        }
        part += __shfl_xor(part, 1, 32);
        part += __shfl_xor(part, 2, 32);
        if (nq == 0)
            y[(size_t)(bb * 2048 + t) * 1024 + head * 64 + p] = part + Dh * xp;
        __syncthreads();
    }
}

// -------------------- gated RMSNorm: rmsnorm(y * silu(z)) * w --------------

__global__ __launch_bounds__(256)
void k_gate_rmsnorm(const float* __restrict__ y, const float* __restrict__ zx,
                    const float* __restrict__ w, __bf16* __restrict__ out) {
    const int row = blockIdx.x;
    const float* yr = y  + (size_t)row * 1024;
    const float* zr = zx + (size_t)row * 2192;   // z = cols [0,1024)
    float g[4];
    float ss = 0.0f;
    #pragma unroll
    for (int j = 0; j < 4; ++j) {
        int c = threadIdx.x + j * 256;
        float v = yr[c] * silu_f(zr[c]);
        g[j] = v;
        ss += v * v;
    }
    ss = block_reduce_sum256(ss);
    float rs = rsqrtf(ss * (1.0f / 1024.0f) + 1e-5f);
    #pragma unroll
    for (int j = 0; j < 4; ++j) {
        int c = threadIdx.x + j * 256;
        out[(size_t)row * 1024 + c] = (__bf16)(g[j] * rs * w[c]);
    }
}

// --------------------------------- launch ----------------------------------

extern "C" void kernel_launch(void* const* d_in, const int* in_sizes, int n_in,
                              void* d_out, int out_size, void* d_ws, size_t ws_size,
                              hipStream_t stream) {
    // setup_inputs() dict order
    const float* x       = (const float*)d_in[0];
    const float* norm_w  = (const float*)d_in[1];
    const float* proj_w  = (const float*)d_in[2];
    const float* proj_b  = (const float*)d_in[3];
    const float* in_w [2] = { (const float*)d_in[4],  (const float*)d_in[12] };
    const float* conv_w[2]= { (const float*)d_in[5],  (const float*)d_in[13] };
    const float* conv_b[2]= { (const float*)d_in[6],  (const float*)d_in[14] };
    const float* dt_b [2] = { (const float*)d_in[7],  (const float*)d_in[15] };
    const float* A_log[2] = { (const float*)d_in[8],  (const float*)d_in[16] };
    const float* Dp   [2] = { (const float*)d_in[9],  (const float*)d_in[17] };
    const float* mnw  [2] = { (const float*)d_in[10], (const float*)d_in[18] };
    const float* out_w[2] = { (const float*)d_in[11], (const float*)d_in[19] };
    float* out = (float*)d_out;

    char* ws = (char*)d_ws;
    const size_t OFF_INW_F  = 0;                         // 2304*512 bf16
    const size_t OFF_INW_B  = OFF_INW_F  + 2304ull*512*2;
    const size_t OFF_OUTW_F = OFF_INW_B  + 2304ull*512*2;   // 512*1024 bf16
    const size_t OFF_OUTW_B = OFF_OUTW_F + 512ull*1024*2;
    const size_t OFF_PROJW  = OFF_OUTW_B + 512ull*1024*2;   // 512*1024 bf16
    const size_t OFF_XN     = OFF_PROJW  + 512ull*1024*2;   // 4096*512 bf16
    const size_t OFF_XNREV  = OFF_XN     + 4096ull*512*2;
    const size_t OFF_ZX     = OFF_XNREV  + 4096ull*512*2;   // 4096*2192 f32
    const size_t OFF_XBC    = OFF_ZX     + 4096ull*2192*4;  // 4096*1152 f32
    const size_t OFF_Y      = OFF_XBC    + 4096ull*1152*4;  // 4096*1024 f32
    const size_t OFF_YG     = OFF_Y      + 4096ull*1024*4;  // 4096*1024 bf16
    const size_t OFF_YCAT   = OFF_YG     + 4096ull*1024*2;  // 4096*1024 bf16

    __bf16* inw_bf [2] = { (__bf16*)(ws + OFF_INW_F),  (__bf16*)(ws + OFF_INW_B) };
    __bf16* outw_bf[2] = { (__bf16*)(ws + OFF_OUTW_F), (__bf16*)(ws + OFF_OUTW_B) };
    __bf16* projw_bf   = (__bf16*)(ws + OFF_PROJW);
    __bf16* xn         = (__bf16*)(ws + OFF_XN);
    __bf16* xn_rev     = (__bf16*)(ws + OFF_XNREV);
    float*  zx         = (float*) (ws + OFF_ZX);
    float*  xBC        = (float*) (ws + OFF_XBC);
    float*  y          = (float*) (ws + OFF_Y);
    __bf16* yg         = (__bf16*)(ws + OFF_YG);
    __bf16* ycat       = (__bf16*)(ws + OFF_YCAT);

    // --- weight conversion (in_w padded 2192 -> 2304 rows) ---
    {
        int tot = 2304 * 512;
        k_cvt_pad_bf16<<<(tot + 255) / 256, 256, 0, stream>>>(in_w[0], inw_bf[0], 2192, 2304, 512);
        k_cvt_pad_bf16<<<(tot + 255) / 256, 256, 0, stream>>>(in_w[1], inw_bf[1], 2192, 2304, 512);
        tot = 512 * 1024;
        k_cvt_pad_bf16<<<(tot + 255) / 256, 256, 0, stream>>>(out_w[0], outw_bf[0], 512, 512, 1024);
        k_cvt_pad_bf16<<<(tot + 255) / 256, 256, 0, stream>>>(out_w[1], outw_bf[1], 512, 512, 1024);
        k_cvt_pad_bf16<<<(tot + 255) / 256, 256, 0, stream>>>(proj_w, projw_bf, 512, 512, 1024);
    }

    // --- input RMSNorm (fwd + flipped) ---
    k_rmsnorm_in<<<4096, 256, 0, stream>>>(x, norm_w, xn, xn_rev);

    // --- two mamba directions (bwd runs entirely in flipped-time domain) ---
    for (int dir = 0; dir < 2; ++dir) {
        const __bf16* a_in = dir ? xn_rev : xn;
        // in-proj: (4096x512) @ (2192x512)^T -> zx
        k_gemm_wmma<0><<<dim3(18, 32), 256, 0, stream>>>(
            a_in, inw_bf[dir], 4096, 2192, 512, 2304, zx, nullptr, 2192, 0, 0,
            nullptr, nullptr);
        // conv + silu
        k_conv_silu<<<(4096 * 1152 + 255) / 256, 256, 0, stream>>>(
            zx, conv_w[dir], conv_b[dir], xBC);
        // selective scan
        k_scan<<<32, 256, 0, stream>>>(zx, xBC, y, dt_b[dir], A_log[dir], Dp[dir]);
        // gate + rmsnorm -> bf16
        k_gate_rmsnorm<<<4096, 256, 0, stream>>>(y, zx, mnw[dir], yg);
        // out-proj: (4096x1024) @ (512x1024)^T -> ycat (bwd: unflip rows, col 512..)
        k_gemm_wmma<1><<<dim3(4, 32), 256, 0, stream>>>(
            yg, outw_bf[dir], 4096, 512, 1024, 512, nullptr, ycat, 1024,
            dir ? 512 : 0, dir ? 1 : 0, nullptr, nullptr);
    }

    // --- final projection with fused bias + residual ---
    k_gemm_wmma<2><<<dim3(4, 32), 256, 0, stream>>>(
        ycat, projw_bf, 4096, 512, 1024, 512, out, nullptr, 512, 0, 0,
        proj_b, x);
}
